// CombinedGNN_31954556682965
// MI455X (gfx1250) — matrix operand order repaired and checked by reading, not compiled, not run on hardware
//
#include <hip/hip_runtime.h>
#include <hip/hip_bf16.h>

typedef __attribute__((ext_vector_type(16))) __bf16 v16bf;
typedef __attribute__((ext_vector_type(8)))  float  v8f;
typedef __attribute__((ext_vector_type(4)))  float  f4;

#define NND   5000      // nodes
#define TT    12
#define DAYN  8
#define FF    7
#define DIMN  96        // T*DAY
#define KPAD  5024      // 157*32  (K padded for bf16 WMMA k=32)
#define NKT   157
#define NCT   7         // 7 col tiles of 16 -> 112 cols (96 feats + deg + pad)
#define COLS  112
#define NMT   313       // row tiles of 16 -> 5008 rows
#define MPAD  5008
#define KCH   8         // K-split chunks for parallelism
#define CHKT  20        // k-tiles per chunk (last chunk: 17)

// ---------------------------------------------------------------------------
// A-tile loader: fp32 source row -> bf16 A operand (16x32), this code's contiguous k order:
// lanes 0-15 hold K = kb+{0..15}, lanes 16-31 hold K = kb+{16..31}
// ---------------------------------------------------------------------------
__device__ inline v16bf load_a_tile(const float* __restrict__ rowptr, int kb,
                                    int kmax, int half) {
  v16bf a;
  const int k0 = kb + half * 16;
  const int k1 = kb + half * 16 + 8;
  if (k1 + 8 <= kmax) {                     // fast vector path
    const f4* p0 = (const f4*)(rowptr + k0);
    const f4* p1 = (const f4*)(rowptr + k1);
    f4 x0 = p0[0], x1 = p0[1], y0 = p1[0], y1 = p1[1];
#pragma unroll
    for (int j = 0; j < 4; ++j) {
      a[j]      = (__bf16)x0[j];
      a[4 + j]  = (__bf16)x1[j];
      a[8 + j]  = (__bf16)y0[j];
      a[12 + j] = (__bf16)y1[j];
    }
  } else {                                   // guarded tail (last K tile)
#pragma unroll
    for (int j = 0; j < 8; ++j) {
      int ka = k0 + j, kc = k1 + j;
      a[j]     = (__bf16)((ka < kmax) ? rowptr[ka] : 0.0f);
      a[8 + j] = (__bf16)((kc < kmax) ? rowptr[kc] : 0.0f);
    }
  }
  return a;
}

// ---------------------------------------------------------------------------
// Kernel 1: pack D' = [data^T | ones | 0-pad] (KPAD x 112) into per-lane bf16
// B layout: tile (ct,kt): lane holds 16 contiguous bf16; element j maps to
// k = kt*32 + (lane<16?0:16) + j , n = ct*16 + (lane&15)
// ---------------------------------------------------------------------------
__global__ void k_pack_b(const float* __restrict__ data, __bf16* __restrict__ bp) {
  int idx = blockIdx.x * blockDim.x + threadIdx.x;
  if (idx >= NCT * NKT * 32) return;
  int ct   = idx / (NKT * 32);
  int rem  = idx % (NKT * 32);
  int kt   = rem >> 5;
  int lane = rem & 31;
  int nb   = ct * 16 + (lane & 15);
  int kb   = kt * 32 + ((lane & 16) ? 16 : 0);
  __bf16* dst = bp + (size_t)idx * 16;
#pragma unroll
  for (int j = 0; j < 16; ++j) {
    int k = kb + j;                       // node index (K dim)
    float v = 0.0f;
    if (k < NND) {
      if (nb < DIMN) {
        int t = nb >> 3, d = nb & 7;
        v = data[((size_t)t * NND + k) * DAYN + d];
      } else if (nb == DIMN) {
        v = 1.0f;                         // ones column -> row degree
      }
    }
    dst[j] = (__bf16)v;
  }
}

// ---------------------------------------------------------------------------
// Kernel 2: partial AGG = adj(chunk of K) @ D'  via v_wmma_f32_16x16x32_bf16.
// grid = (40 row-macro-tiles, 8 K-chunks); 8 waves/block, 16-row strip/wave.
// B tiles (7KB per k-step, shared by all 8 waves) are double-buffered in LDS
// via global_load_async_to_lds_b128 (ASYNCcnt).  B tiles are consumed in two
// pinned groups (4 + 3) -> batched ds_loads with 2 dscnt waits per k-step,
// and peak register pressure stays under the spill threshold.
// ---------------------------------------------------------------------------
__global__ void k_gemm_agg(const float* __restrict__ adj,
                           const __bf16* __restrict__ bp,
                           float* __restrict__ aggp) {
  __shared__ __align__(16) __bf16 sB[2 * NCT * 512];   // 2 x 7168 B

  const int tid  = threadIdx.x;
  const int wave = tid >> 5;
  const int lane = tid & 31;
  const int ch   = blockIdx.y;
  const int kt0  = ch * CHKT;
  const int kt1  = (kt0 + CHKT < NKT) ? (kt0 + CHKT) : NKT;
  const int tt   = blockIdx.x * 8 + wave;
  const bool active = (tt < NMT);

  const int m    = lane & 15;
  const int half = (lane >> 4) & 1;
  int row  = tt * 16 + m;
  int rowc = (row < NND) ? row : NND - 1;        // clamp pad rows
  const float* rowptr = adj + (size_t)rowc * NND;

  // async B staging: threads 0..223 each own one 32-byte segment (ct,lane)
  const bool stager = (tid < NCT * 32);
  const uint32_t lds_base = (uint32_t)(uintptr_t)(void*)sB;   // LDS offset
  const char* gseg = (const char*)bp +
      ((size_t)((tid >> 5) * NKT + kt0) * 32 + (tid & 31)) * 32;

  v8f acc[NCT];
#pragma unroll
  for (int c = 0; c < NCT; ++c)
#pragma unroll
    for (int j = 0; j < 8; ++j) acc[c][j] = 0.0f;

  // prologue: stage kt0 into buffer 0
  if (stager) {
    uint32_t l = lds_base + (uint32_t)tid * 32;
    const char* g = gseg;
    asm volatile("global_load_async_to_lds_b128 %0, %1, off"
                 :: "v"(l), "v"(g) : "memory");
    asm volatile("global_load_async_to_lds_b128 %0, %1, off offset:16"
                 :: "v"(l + 16), "v"(g) : "memory");
  }
  asm volatile("s_wait_asynccnt 0x0" ::: "memory");
  __syncthreads();

  for (int kt = kt0; kt < kt1; ++kt) {
    const int buf = (kt - kt0) & 1;
    // stage next k-step into the other buffer while computing this one
    if (kt + 1 < kt1 && stager) {
      uint32_t l = lds_base + (uint32_t)((buf ^ 1) * 7168 + tid * 32);
      const char* g = gseg + (size_t)(kt + 1 - kt0) * 1024;
      asm volatile("global_load_async_to_lds_b128 %0, %1, off"
                   :: "v"(l), "v"(g) : "memory");
      asm volatile("global_load_async_to_lds_b128 %0, %1, off offset:16"
                   :: "v"(l + 16), "v"(g) : "memory");
    }
    if (active) {
      int kb = kt * 32;
      if (kb + 64 < NND) __builtin_prefetch(rowptr + kb + 32, 0, 1);
      v16bf a = load_a_tile(rowptr, kb, NND, half);
      const __bf16* lb = &sB[(size_t)buf * 3584 + (size_t)lane * 16];
      // group 1: tiles 0..3 (8 ds_load_b128, one wait, 4 WMMAs)
      v16bf b0 = *(const v16bf*)(lb + 0 * 512);
      v16bf b1 = *(const v16bf*)(lb + 1 * 512);
      v16bf b2 = *(const v16bf*)(lb + 2 * 512);
      v16bf b3 = *(const v16bf*)(lb + 3 * 512);
      asm("" : "+v"(b0), "+v"(b1), "+v"(b2), "+v"(b3));
      acc[0] = __builtin_amdgcn_wmma_f32_16x16x32_bf16(
          false, a, false, b0, (short)0, acc[0], false, false);
      acc[1] = __builtin_amdgcn_wmma_f32_16x16x32_bf16(
          false, a, false, b1, (short)0, acc[1], false, false);
      acc[2] = __builtin_amdgcn_wmma_f32_16x16x32_bf16(
          false, a, false, b2, (short)0, acc[2], false, false);
      acc[3] = __builtin_amdgcn_wmma_f32_16x16x32_bf16(
          false, a, false, b3, (short)0, acc[3], false, false);
      // group 2: tiles 4..6 (6 ds_load_b128, one wait, 3 WMMAs)
      v16bf b4 = *(const v16bf*)(lb + 4 * 512);
      v16bf b5 = *(const v16bf*)(lb + 5 * 512);
      v16bf b6 = *(const v16bf*)(lb + 6 * 512);
      asm("" : "+v"(b4), "+v"(b5), "+v"(b6));
      acc[4] = __builtin_amdgcn_wmma_f32_16x16x32_bf16(
          false, a, false, b4, (short)0, acc[4], false, false);
      acc[5] = __builtin_amdgcn_wmma_f32_16x16x32_bf16(
          false, a, false, b5, (short)0, acc[5], false, false);
      acc[6] = __builtin_amdgcn_wmma_f32_16x16x32_bf16(
          false, a, false, b6, (short)0, acc[6], false, false);
    }
    asm volatile("s_wait_asynccnt 0x0" ::: "memory");
    __syncthreads();
  }

  // C layout: col = lane&15, VGPR v -> row v + (lane>=16 ? 8 : 0)
  if (active) {
    float* dst = aggp + (size_t)ch * MPAD * COLS;
#pragma unroll
    for (int v = 0; v < 8; ++v) {
      int r = tt * 16 + v + half * 8;
      if (r < NND) {
#pragma unroll
        for (int c = 0; c < NCT; ++c)
          dst[(size_t)r * COLS + c * 16 + m] = acc[c][v];
      }
    }
  }
}

// ---------------------------------------------------------------------------
// Kernel 2b: reduce the 8 K-chunk partials into the final AGG buffer.
// ---------------------------------------------------------------------------
__global__ void k_reduce(const float* __restrict__ aggp, float* __restrict__ agg) {
  size_t i = (size_t)blockIdx.x * blockDim.x + threadIdx.x;
  if (i >= (size_t)NND * COLS) return;
  float s = 0.0f;
#pragma unroll
  for (int ch = 0; ch < KCH; ++ch) s += aggp[(size_t)ch * MPAD * COLS + i];
  agg[i] = s;
}

// ---------------------------------------------------------------------------
// Kernel 3: per-node 12-step recurrence (row-independent). Weights in LDS.
// masks[1] == 0 in the reference (elementwise power) -> third segment skipped.
// ---------------------------------------------------------------------------
__global__ void k_recur(const float* __restrict__ data, const float* __restrict__ pos,
                        const float* __restrict__ hisW, const float* __restrict__ curW,
                        const float* __restrict__ hisw, const float* __restrict__ curw,
                        const float* __restrict__ agg, float* __restrict__ fin) {
  __shared__ float sW[3077];
  float* s_hisW = sW;            // 12*7*28 = 2352
  float* s_curW = sW + 2352;     // 12*4    = 48
  float* s_hw   = sW + 2400;     // 7*95    = 665
  float* s_cw   = sW + 3065;     // 12
  for (int i = threadIdx.x; i < 2352; i += blockDim.x) s_hisW[i] = hisW[i];
  for (int i = threadIdx.x; i < 48;   i += blockDim.x) s_curW[i] = curW[i];
  for (int i = threadIdx.x; i < 665;  i += blockDim.x) s_hw[i]   = hisw[i];
  for (int i = threadIdx.x; i < 12;   i += blockDim.x) s_cw[i]   = curw[i];
  __syncthreads();

  int n = blockIdx.x * blockDim.x + threadIdx.x;
  if (n >= NND) return;

  float deg = agg[(size_t)n * COLS + DIMN];
  float inv = deg > 0.0f ? 1.0f / deg : 1.0f;

  float hist[TT * FF];
  float curh[TT];
  float hprev[FF];
#pragma unroll
  for (int i = 0; i < FF; ++i) hprev[i] = 0.0f;
  float cprev = 0.0f;

#pragma unroll
  for (int t = 0; t < TT; ++t) {
    const float* drow = data + ((size_t)t * NND + n) * DAYN;
    const float* prow = pos  + ((size_t)t * NND + n) * DAYN;
    float raw[DAYN], ag[DAYN];
#pragma unroll
    for (int d = 0; d < DAYN; ++d) {
      raw[d] = drow[d];
      ag[d]  = agg[(size_t)n * COLS + t * DAYN + d] * inv;
    }
    // his: combined = [raw | agg | 0 | hprev] @ his_W[t].T, relu, +pos
#pragma unroll
    for (int i = 0; i < FF; ++i) {
      const float* w = s_hisW + (t * FF + i) * 28;
      float s = 0.0f;
#pragma unroll
      for (int j = 0; j < FF; ++j)
        s += raw[j] * w[j] + ag[j] * w[7 + j] + hprev[j] * w[21 + j];
      s = s > 0.0f ? s : 0.0f;
      hist[t * FF + i] = s + prow[i];
    }
#pragma unroll
    for (int i = 0; i < FF; ++i) {
      float s = 0.0f;
#pragma unroll
      for (int j = 0; j < (t + 1) * FF; ++j) s += hist[j] * s_hw[i * 95 + j];
      hprev[i] = s > 0.0f ? s : 0.0f;
    }
    // cur
    {
      const float* w = s_curW + t * 4;
      float s = raw[7] * w[0] + ag[7] * w[1] + cprev * w[3];
      s = s > 0.0f ? s : 0.0f;
      curh[t] = s + prow[7];
      float s2 = 0.0f;
#pragma unroll
      for (int j = 0; j <= t; ++j) s2 += curh[j] * s_cw[j];
      cprev = s2 > 0.0f ? s2 : 0.0f;
    }
  }
  float* fr = fin + (size_t)n * DIMN;
#pragma unroll
  for (int i = 0; i < TT * FF; ++i) fr[i] = hist[i];
#pragma unroll
  for (int i = 0; i < TT; ++i) fr[84 + i] = curh[i];
}

// ---------------------------------------------------------------------------
// Kernel 4: OUT = relu(FIN(5000x96) @ final_weight(96x96).T)  via bf16 WMMA.
// B[k][n] = Wf[n][k] -> per-lane 16 contiguous fp32 from row n of Wf.
// ---------------------------------------------------------------------------
__global__ void k_gemm_final(const float* __restrict__ fin,
                             const float* __restrict__ wf,
                             float* __restrict__ out) {
  int wave = threadIdx.x >> 5;
  int lane = threadIdx.x & 31;
  int tt   = blockIdx.x * 8 + wave;
  if (tt >= NMT) return;
  int m    = lane & 15;
  int half = (lane >> 4) & 1;
  int row  = tt * 16 + m;
  int rowc = row < NND ? row : NND - 1;
  const float* rowptr = fin + (size_t)rowc * DIMN;

  v8f acc[6];
#pragma unroll
  for (int c = 0; c < 6; ++c)
#pragma unroll
    for (int j = 0; j < 8; ++j) acc[c][j] = 0.0f;

#pragma unroll
  for (int kt = 0; kt < 3; ++kt) {
    v16bf a = load_a_tile(rowptr, kt * 32, DIMN, half);
    v16bf bt[6];
#pragma unroll
    for (int c = 0; c < 6; ++c) {
      const float* wrow = wf + (size_t)(c * 16 + m) * DIMN + kt * 32 + half * 16;
      const f4* wp = (const f4*)wrow;
      f4 w0 = wp[0], w1 = wp[1], w2 = wp[2], w3 = wp[3];
#pragma unroll
      for (int j = 0; j < 4; ++j) {
        bt[c][j]      = (__bf16)w0[j];
        bt[c][4 + j]  = (__bf16)w1[j];
        bt[c][8 + j]  = (__bf16)w2[j];
        bt[c][12 + j] = (__bf16)w3[j];
      }
    }
#pragma unroll
    for (int c = 0; c < 6; ++c)
      acc[c] = __builtin_amdgcn_wmma_f32_16x16x32_bf16(
          false, a, false, bt[c], (short)0, acc[c], false, false);
  }
#pragma unroll
  for (int v = 0; v < 8; ++v) {
    int r = tt * 16 + v + half * 8;
    if (r < NND) {
#pragma unroll
      for (int c = 0; c < 6; ++c) {
        float x = acc[c][v];
        out[(size_t)r * DIMN + c * 16 + m] = x > 0.0f ? x : 0.0f;
      }
    }
  }
}

// ---------------------------------------------------------------------------
extern "C" void kernel_launch(void* const* d_in, const int* in_sizes, int n_in,
                              void* d_out, int out_size, void* d_ws, size_t ws_size,
                              hipStream_t stream) {
  const float* adj  = (const float*)d_in[0];
  const float* data = (const float*)d_in[1];
  const float* pos  = (const float*)d_in[2];
  const float* hisW = (const float*)d_in[3];
  const float* curW = (const float*)d_in[4];
  const float* hisw = (const float*)d_in[5];
  const float* curw = (const float*)d_in[6];
  const float* wf   = (const float*)d_in[7];
  float* out = (float*)d_out;

  char* ws = (char*)d_ws;
  const size_t bpack_bytes = (size_t)NCT * NKT * 32 * 16 * 2;        // ~1.07 MB
  const size_t aggp_bytes  = (size_t)KCH * MPAD * COLS * 4;          // ~17.9 MB
  const size_t agg_bytes   = (size_t)MPAD * COLS * 4;                // ~2.24 MB
  size_t off0 = 0;
  size_t off1 = (off0 + bpack_bytes + 255) & ~(size_t)255;
  size_t off2 = (off1 + aggp_bytes + 255) & ~(size_t)255;
  size_t off3 = (off2 + agg_bytes + 255) & ~(size_t)255;
  __bf16* bpack = (__bf16*)(ws + off0);
  float*  aggp  = (float*)(ws + off1);   // KCH partial AGG buffers
  float*  agg   = (float*)(ws + off2);   // reduced AGG
  float*  fin   = (float*)(ws + off3);   // 5008*96 fp32

  int packN = NCT * NKT * 32;                                        // 35168
  k_pack_b<<<(packN + 255) / 256, 256, 0, stream>>>(data, bpack);
  k_gemm_agg<<<dim3((NMT + 7) / 8, KCH), 256, 0, stream>>>(adj, bpack, aggp);
  size_t redN = (size_t)NND * COLS;
  k_reduce<<<(int)((redN + 255) / 256), 256, 0, stream>>>(aggp, agg);
  k_recur<<<(NND + 255) / 256, 256, 0, stream>>>(data, pos, hisW, curW,
                                                 hisw, curw, agg, fin);
  k_gemm_final<<<(NMT + 7) / 8, 256, 0, stream>>>(fin, wf, out);
}
